// GraphFunction_19499151524398
// MI455X (gfx1250) — matrix operand order repaired and checked by reading, not compiled, not run on hardware
//
#include <hip/hip_runtime.h>
#include <math.h>

typedef __attribute__((ext_vector_type(16))) _Float16 v16h;
typedef __attribute__((ext_vector_type(8)))  _Float16 v8h;
typedef __attribute__((ext_vector_type(4)))  _Float16 v4h;
typedef __attribute__((ext_vector_type(8)))  float    v8f;

#define FDIM 128
#define HDIM 128
#define DDIM 16

// ---------------- degree / symmetric normalization ----------------

__global__ void k_init_deg(float* __restrict__ deg, int n) {
  int i = blockIdx.x * blockDim.x + threadIdx.x;
  if (i < n) deg[i] = 1.0f;                       // self-loop weight
}

__global__ void k_scatter_deg(const int* __restrict__ col, const float* __restrict__ w,
                              float* __restrict__ deg, int e) {
  int i = blockIdx.x * blockDim.x + threadIdx.x;
  if (i < e) atomicAdd(&deg[col[i]], w[i]);
}

__global__ void k_dis(float* __restrict__ deg, int n) {
  int i = blockIdx.x * blockDim.x + threadIdx.x;
  if (i < n) {
    float d = deg[i];
    deg[i] = (d > 0.0f) ? rsqrtf(d) : 0.0f;       // dis = deg^-1/2
  }
}

__global__ void k_norm(const int* __restrict__ row, const int* __restrict__ col,
                       const float* __restrict__ w, const float* __restrict__ dis,
                       float* __restrict__ nrm, int e) {
  int i = blockIdx.x * blockDim.x + threadIdx.x;
  if (i < e) nrm[i] = dis[row[i]] * w[i] * dis[col[i]];
}

// ---------------- operand packing ----------------

// f32 -> f16 elementwise, 4 at a time (layer-1 input conversion)
__global__ void k_pack_x(const float* __restrict__ X, _Float16* __restrict__ Xh, long long total4) {
  long long idx = (long long)blockIdx.x * blockDim.x + threadIdx.x;
  if (idx >= total4) return;
  float4 v = ((const float4*)X)[idx];
  v4h o; o[0] = (_Float16)v.x; o[1] = (_Float16)v.y; o[2] = (_Float16)v.z; o[3] = (_Float16)v.w;
  ((v4h*)Xh)[idx] = o;
}

// W[K x nout] f32 row-major -> packed f16 B-fragments:
// Wp[((k/32)*nout + col)*32 + (k%32)] so each lane's 16 B values are two contiguous runs of 8.
template <int K, int NOUT>
__global__ void k_pack_w(const float* __restrict__ W, _Float16* __restrict__ Wp) {
  int idx = blockIdx.x * blockDim.x + threadIdx.x;
  if (idx >= K * NOUT) return;
  int k = idx / NOUT;
  int c = idx - k * NOUT;
  Wp[((k >> 5) * NOUT + c) * 32 + (k & 31)] = (_Float16)W[idx];
}

// ---------------- WMMA GEMM ----------------
// Hout[nrows x NOUT] = Xh[nrows x K] @ W[K x NOUT]  (f16 in, f32 accumulate)
// Block = (NOUT/16) waves; wave w owns column tile w; A strip staged in LDS, shared by all waves.
template <int K, int NOUT>
__global__ void k_gemm_wmma(const _Float16* __restrict__ Xh, const _Float16* __restrict__ Wp,
                            float* __restrict__ Hout, int nrows) {
  __shared__ __align__(16) _Float16 As[16][K + 8];  // padded: row stride 68 dwords -> conflict-free

  const int row0 = blockIdx.x * 16;
  const int tid  = threadIdx.x;
  const int wave = tid >> 5;
  const int lane = tid & 31;
  const int half = lane >> 4;       // 0: lanes 0-15, 1: lanes 16-31
  const int mn   = lane & 15;
  const int col0 = wave * 16;

  // cooperative stage of the 16 x K f16 A strip into LDS (16B per transfer)
  constexpr int NSEG = K >> 3;                    // 8-f16 segments per row
  for (int s = tid; s < 16 * NSEG; s += 32 * (NOUT / 16)) {
    int r = s / NSEG, seg = s - r * NSEG;
    int gr = row0 + r; if (gr >= nrows) gr = nrows - 1;   // clamp (keeps later EXEC all-1s)
    *(float4*)&As[r][seg * 8] = *(const float4*)(Xh + (size_t)gr * K + seg * 8);
  }
  __syncthreads();

  v8f acc = {};
  #pragma unroll
  for (int k0 = 0; k0 < K; k0 += 32) {
    // A 16x32 f16 fragment from LDS: elems 0-7 -> K = k0+8*half+i ; elems 8-15 -> K = k0+16+8*half+(i-8)
    union { v16h v; struct { v8h lo, hi; } p; } au;
    au.p.lo = *(const v8h*)&As[mn][k0 + half * 8];
    au.p.hi = *(const v8h*)&As[mn][k0 + 16 + half * 8];
    // B 32x16 f16 fragment: lane holds column col0+mn; elems 0-7 -> K = k0+8*half+i ; elems 8-15 -> K = k0+16+8*half+(i-8)
    const _Float16* bp = Wp + ((k0 >> 5) * NOUT + col0 + mn) * 32 + half * 8;
    union { v16h v; struct { v8h lo, hi; } p; } bu;
    bu.p.lo = *(const v8h*)bp;
    bu.p.hi = *(const v8h*)(bp + 16);
    acc = __builtin_amdgcn_wmma_f32_16x16x32_f16(
        /*neg_a=*/false, au.v, /*neg_b=*/false, bu.v,
        /*c_mod=*/(short)0, acc, /*reuse_a=*/false, /*reuse_b=*/false);
  }

  // D f32 16x16: lane N = mn; VGPR r -> M = r + 8*half
  if (row0 + 16 <= nrows) {                       // common path: full tile, unguarded stores
    float* __restrict__ orow = Hout + (size_t)(row0 + half * 8) * NOUT + col0 + mn;
    #pragma unroll
    for (int r = 0; r < 8; ++r) orow[r * NOUT] = acc[r];   // constant immediate offsets
  } else {
    #pragma unroll
    for (int r = 0; r < 8; ++r) {
      int orow = row0 + half * 8 + r;
      if (orow < nrows) Hout[(size_t)orow * NOUT + col0 + mn] = acc[r];
    }
  }
}

// ---------------- propagation: agg[c] = dis[c]^2 * h[c] + sum_e norm[e]*h[row[e]] ----------------

template <int HD>
__global__ void k_selfloop(const float* __restrict__ dis, const float* __restrict__ h,
                           float* __restrict__ agg, long long total) {
  constexpr int HQ = HD / 4;
  long long idx = (long long)blockIdx.x * blockDim.x + threadIdx.x;
  if (idx >= total) return;
  int i = (int)(idx / HQ);                        // power-of-2 -> shift
  int q = (int)(idx - (long long)i * HQ);
  float s = dis[i];
  s = s * s;                                      // self-loop norm = dis*1*dis
  const float4 hv = *(const float4*)(h + (size_t)i * HD + q * 4);
  float4 o; o.x = s * hv.x; o.y = s * hv.y; o.z = s * hv.z; o.w = s * hv.w;
  *(float4*)(agg + (size_t)i * HD + q * 4) = o;   // full overwrite -> no memset needed
}

template <int HD>
__global__ void k_scatter_edges(const int* __restrict__ row, const int* __restrict__ col,
                                const float* __restrict__ nrm, const float* __restrict__ h,
                                float* __restrict__ agg, long long total) {
  constexpr int HQ = HD / 4;
  long long idx = (long long)blockIdx.x * blockDim.x + threadIdx.x;
  if (idx >= total) return;
  int e = (int)(idx / HQ);                        // power-of-2 -> shift
  int q = (int)(idx - (long long)e * HQ);
  int r = row[e], c = col[e];
  float nv = nrm[e];
  const float4 hv = *(const float4*)(h + (size_t)r * HD + q * 4);
  float* dst = agg + (size_t)c * HD + q * 4;
  atomicAdd(dst + 0, nv * hv.x);
  atomicAdd(dst + 1, nv * hv.y);
  atomicAdd(dst + 2, nv * hv.z);
  atomicAdd(dst + 3, nv * hv.w);
}

// bias + sigmoid, writing next layer's GEMM input directly as f16
template <int HD>
__global__ void k_bias_sigmoid_f16(const float* __restrict__ agg, const float* __restrict__ bias,
                                   _Float16* __restrict__ xo, long long total) {
  long long idx = (long long)blockIdx.x * blockDim.x + threadIdx.x;
  if (idx >= total) return;
  int f = (int)(idx & (HD - 1));                  // HD is a power of 2
  float v = agg[idx] + bias[f];
  xo[idx] = (_Float16)(1.0f / (1.0f + __expf(-v)));
}

__global__ void k_out_transpose(const float* __restrict__ agg, const float* __restrict__ b3,
                                float* __restrict__ out, int n) {
  int idx = blockIdx.x * blockDim.x + threadIdx.x;
  if (idx >= n * DDIM) return;
  int i = idx >> 4;          // node
  int d = idx & 15;          // output dim
  out[(size_t)d * n + i] = agg[idx] + b3[d];      // bias + transpose to [D, N]
}

// ---------------- driver ----------------

extern "C" void kernel_launch(void* const* d_in, const int* in_sizes, int n_in,
                              void* d_out, int out_size, void* d_ws, size_t ws_size,
                              hipStream_t stream) {
  const float* x0  = (const float*)d_in[0];   // [N, 128]
  const int*   ei  = (const int*)d_in[1];     // [2, E]
  const float* ew  = (const float*)d_in[2];   // [E]
  const float* W1  = (const float*)d_in[3];
  const float* b1  = (const float*)d_in[4];
  const float* W2  = (const float*)d_in[5];
  const float* b2  = (const float*)d_in[6];
  const float* W3  = (const float*)d_in[7];
  const float* b3  = (const float*)d_in[8];
  float* out = (float*)d_out;

  const int N = in_sizes[0] / FDIM;
  const int E = in_sizes[1] / 2;
  const int* row = ei;
  const int* col = ei + E;

  // carve workspace: dis + norm + Xh(f16) + Wp(f16) + h + agg  ~ 141 MB
  char* ws = (char*)d_ws;
  auto carve = [&](size_t bytes) -> char* {
    char* p = ws;
    ws += (bytes + 255) & ~(size_t)255;
    return p;
  };
  float*    dis  = (float*)   carve((size_t)N * sizeof(float));       // deg -> rsqrt(deg) in place
  float*    nrm  = (float*)   carve((size_t)E * sizeof(float));
  _Float16* Xh   = (_Float16*)carve((size_t)N * HDIM * sizeof(_Float16));
  _Float16* Wp   = (_Float16*)carve((size_t)FDIM * HDIM * sizeof(_Float16));
  float*    hbuf = (float*)   carve((size_t)N * HDIM * sizeof(float));
  float*    agg  = (float*)   carve((size_t)N * HDIM * sizeof(float));
  (void)ws_size; (void)n_in; (void)out_size;

  const int T = 256;
  // --- normalization coefficients ---
  k_init_deg<<<(N + T - 1) / T, T, 0, stream>>>(dis, N);
  k_scatter_deg<<<(E + T - 1) / T, T, 0, stream>>>(col, ew, dis, E);
  k_dis<<<(N + T - 1) / T, T, 0, stream>>>(dis, N);
  k_norm<<<(E + T - 1) / T, T, 0, stream>>>(row, col, ew, dis, nrm, E);

  const int rowTiles = (N + 15) / 16;
  const long long totH  = (long long)N * (HDIM / 4);
  const long long teH   = (long long)E * (HDIM / 4);
  const long long totHf = (long long)N * HDIM;

  // --- layer 1: sigmoid(conv(x0, W1, b1)) ---
  k_pack_x<<<(int)((totH + T - 1) / T), T, 0, stream>>>(x0, Xh, totH);  // N*128/4 vec4 conversions
  k_pack_w<FDIM, HDIM><<<(FDIM * HDIM + T - 1) / T, T, 0, stream>>>(W1, Wp);
  k_gemm_wmma<FDIM, HDIM><<<rowTiles, 32 * (HDIM / 16), 0, stream>>>(Xh, Wp, hbuf, N);
  k_selfloop<HDIM><<<(int)((totH + T - 1) / T), T, 0, stream>>>(dis, hbuf, agg, totH);
  k_scatter_edges<HDIM><<<(int)((teH + T - 1) / T), T, 0, stream>>>(row, col, nrm, hbuf, agg, teH);
  k_bias_sigmoid_f16<HDIM><<<(int)((totHf + T - 1) / T), T, 0, stream>>>(agg, b1, Xh, totHf);

  // --- layer 2: sigmoid(conv(x1, W2, b2)) ---
  k_pack_w<HDIM, HDIM><<<(HDIM * HDIM + T - 1) / T, T, 0, stream>>>(W2, Wp);
  k_gemm_wmma<HDIM, HDIM><<<rowTiles, 32 * (HDIM / 16), 0, stream>>>(Xh, Wp, hbuf, N);
  k_selfloop<HDIM><<<(int)((totH + T - 1) / T), T, 0, stream>>>(dis, hbuf, agg, totH);
  k_scatter_edges<HDIM><<<(int)((teH + T - 1) / T), T, 0, stream>>>(row, col, nrm, hbuf, agg, teH);
  k_bias_sigmoid_f16<HDIM><<<(int)((totHf + T - 1) / T), T, 0, stream>>>(agg, b2, Xh, totHf);

  // --- layer 3: conv(x2, W3, b3), nout = 16 ---
  const long long totD = (long long)N * (DDIM / 4);
  const long long teD  = (long long)E * (DDIM / 4);
  k_pack_w<HDIM, DDIM><<<(HDIM * DDIM + T - 1) / T, T, 0, stream>>>(W3, Wp);
  k_gemm_wmma<HDIM, DDIM><<<rowTiles, 32 * (DDIM / 16), 0, stream>>>(Xh, Wp, hbuf, N);
  k_selfloop<DDIM><<<(int)((totD + T - 1) / T), T, 0, stream>>>(dis, hbuf, agg, totD);
  k_scatter_edges<DDIM><<<(int)((teD + T - 1) / T), T, 0, stream>>>(row, col, nrm, hbuf, agg, teD);
  k_out_transpose<<<(N * DDIM + T - 1) / T, T, 0, stream>>>(agg, b3, out, N);
}
